// DubinsLSTMEncoderDecoder_2499670966848
// MI455X (gfx1250) — matrix operand order repaired and checked by reading, not compiled, not run on hardware
//
#include <hip/hip_runtime.h>
#include <hip/hip_bf16.h>

// ---------------------------------------------------------------------------
// Persistent-block LSTM encoder/decoder for MI455X (gfx1250, wave32, WMMA).
//   grid = 256 blocks x 256 threads (8 waves). Each block owns 16 batch rows,
//   keeps c in WMMA C-fragment registers, h in LDS (bf16), and loops T steps.
//   Big GEMMs (h@Whh^T, h@Wo1^T, enc@W2^T) use v_wmma_f32_16x16x32_bf16.
// ---------------------------------------------------------------------------

typedef __bf16 bf16_t;
typedef bf16_t v16bf __attribute__((ext_vector_type(16)));
typedef float  v8f   __attribute__((ext_vector_type(8)));
typedef unsigned int v4u __attribute__((ext_vector_type(4)));

union FragAB { v4u u[2]; v16bf v; };

__device__ __forceinline__ unsigned short f2bf(float f) {
    union { float f; unsigned u; } cv; cv.f = f;
    unsigned u = cv.u;
    return (unsigned short)((u + 0x7fffu + ((u >> 16) & 1u)) >> 16);  // RNE
}

__device__ __forceinline__ float sigf(float x) {
    return 1.0f / (1.0f + __expf(-x));
}
__device__ __forceinline__ float tanhfast(float x) {
    // saturates correctly for large |x| (exp -> inf/0)
    return 1.0f - 2.0f / (1.0f + __expf(2.0f * x));
}

__device__ __forceinline__ v8f wmma_bf16(v16bf a, v16bf b, v8f c) {
    return __builtin_amdgcn_wmma_f32_16x16x32_bf16(
        /*neg_a=*/false, a, /*neg_b=*/false, b,
        /*c_mod=*/(short)0, c, /*reuse_a=*/false, /*reuse_b=*/false);
}

// A fragment (16x32 bf16) from LDS row `row` (row = &sh_h[lane&15][0]):
//   lanes 0-15: V0..7 K=K0+0..15
//   lanes16-31: V0..7 K=K0+16..31
__device__ __forceinline__ v16bf load_a(const unsigned short* row, int K0, int lhalf) {
    FragAB f;
    f.u[0] = *(const v4u*)(row + K0 + lhalf * 16);
    f.u[1] = *(const v4u*)(row + K0 + lhalf * 16 + 8);
    return f.v;
}

// B fragment (32x16 bf16), weights row-major [N][K], lane holds col n=lane&15,
// K = (lane>>4)*16 + 0..15 contiguous -> two global_load_b128.
__device__ __forceinline__ v16bf load_b(const unsigned short* base) {
    FragAB f;
    f.u[0] = *(const v4u*)(base);
    f.u[1] = *(const v4u*)(base + 8);
    return f.v;
}

#define HSZ 256
#define N2 (512 * 256)
#define NH (1024 * 256)
#define NO (128 * 256)

// f32 -> bf16 weight packing (runs once per launch, deterministic).
__global__ void conv_bf16_kernel(const float* __restrict__ W2,
                                 const float* __restrict__ Whh,
                                 const float* __restrict__ Wo1,
                                 unsigned short* __restrict__ ws) {
    int i = blockIdx.x * blockDim.x + threadIdx.x;
    if (i < N2)                ws[i] = f2bf(W2[i]);
    else if (i < N2 + NH)      ws[i] = f2bf(Whh[i - N2]);
    else if (i < N2 + NH + NO) ws[i] = f2bf(Wo1[i - N2 - NH]);
}

__global__ __launch_bounds__(256, 1)
void dubins_lstm_kernel(const float* __restrict__ meta,
                        const int*   __restrict__ tslp,
                        const float* __restrict__ W1,  const float* __restrict__ b1,
                        const float* __restrict__ b2,
                        const float* __restrict__ Wih, const float* __restrict__ bih,
                        const float* __restrict__ bhh,
                        const float* __restrict__ bo1, const float* __restrict__ Wo2,
                        const float* __restrict__ bo2,
                        const unsigned short* __restrict__ w2b,
                        const unsigned short* __restrict__ whhb,
                        const unsigned short* __restrict__ wo1b,
                        float* __restrict__ out) {
    const int tid   = threadIdx.x;
    const int wave  = tid >> 5;
    const int lane  = tid & 31;
    const int lmod  = lane & 15;
    const int lhalf = lane >> 4;
    const int row0  = blockIdx.x * 16;
    const int T     = tslp[0];

    __shared__ __align__(16) unsigned short sh_h[16][264];   // h (bf16), padded
    __shared__ float sh_x[16][4];                             // decoder input x
    __shared__ float sh_z1[16][132];                          // relu(h@Wo1^T+bo1)
    __shared__ float sh_wo2[3][128];
    __shared__ float sh_bo2v[4];

    // ---- one-time small staging ----
    for (int e = tid; e < 3 * 128; e += 256) sh_wo2[e >> 7][e & 127] = Wo2[e];
    if (tid < 3)  sh_bo2v[tid] = bo2[tid];
    if (tid < 48) sh_x[tid / 3][tid % 3] = meta[(row0 + tid / 3) * 7 + (tid % 3)];

    // ---- encoder layer 1 (K=7, scalar is optimal) -> sh_h as bf16 ----
    for (int e = tid; e < 16 * 256; e += 256) {
        int m = e >> 8, n = e & 255;
        float acc = b1[n];
#pragma unroll
        for (int p = 0; p < 7; ++p) acc += meta[(row0 + m) * 7 + p] * W1[n * 7 + p];
        sh_h[m][n] = f2bf(fmaxf(acc, 0.0f));
    }
    __syncthreads();

    // ---- encoder layer 2: enc2 = relu(enc1 @ W2^T + b2), 16x512, K=256 ----
    // wave w: tiles {2w,2w+1} -> h0 columns, {16+2w,16+2w+1} -> c0 columns.
    v8f enc[4];
#pragma unroll
    for (int ti = 0; ti < 4; ++ti) {
        int nt = (ti < 2) ? (2 * wave + ti) : (16 + 2 * wave + (ti - 2));
        int nb = nt * 16;
        v8f acc = {0.f, 0.f, 0.f, 0.f, 0.f, 0.f, 0.f, 0.f};
        for (int k0 = 0; k0 < 8; ++k0) {
            int K0 = k0 * 32;
            v16bf a  = load_a(&sh_h[lmod][0], K0, lhalf);
            v16bf bb = load_b(w2b + (nb + lmod) * HSZ + K0 + lhalf * 16);
            acc = wmma_bf16(a, bb, acc);
        }
        float bias = b2[nb + lmod];
#pragma unroll
        for (int r = 0; r < 8; ++r) acc[r] = fmaxf(acc[r] + bias, 0.0f);
        enc[ti] = acc;
    }
    __syncthreads();  // enc1 reads complete before overwriting sh_h with h0

    v8f cst[2] = {enc[2], enc[3]};  // c state stays in C-fragment registers
#pragma unroll
    for (int ti = 0; ti < 2; ++ti) {
        int nb = (2 * wave + ti) * 16;
#pragma unroll
        for (int r = 0; r < 8; ++r)
            sh_h[lhalf * 8 + r][nb + lmod] = f2bf(enc[ti][r]);
    }

    // ---- hoist per-lane constants for gate finishing (n fixed per lane) ----
    float wx[2][4][3], bsum[2][4];
#pragma unroll
    for (int ti = 0; ti < 2; ++ti)
#pragma unroll
        for (int q = 0; q < 4; ++q) {
            int n = q * 256 + (2 * wave + ti) * 16 + lmod;
            wx[ti][q][0] = Wih[n * 3 + 0];
            wx[ti][q][1] = Wih[n * 3 + 1];
            wx[ti][q][2] = Wih[n * 3 + 2];
            bsum[ti][q]  = bih[n] + bhh[n];
        }
    const float bo1v = bo1[wave * 16 + lmod];
    __syncthreads();  // h0 visible to all waves

    // =======================  T sequential LSTM steps  =======================
    for (int t = 0; t < T; ++t) {
        // (1) gates = h @ Whh^T : M=16, N=1024, K=256; 64 WMMA / wave / step
        v8f g[2][4];
#pragma unroll
        for (int ti = 0; ti < 2; ++ti)
#pragma unroll
            for (int q = 0; q < 4; ++q)
                g[ti][q] = (v8f){0.f, 0.f, 0.f, 0.f, 0.f, 0.f, 0.f, 0.f};

        for (int k0 = 0; k0 < 8; ++k0) {
            int K0 = k0 * 32;
            v16bf a = load_a(&sh_h[lmod][0], K0, lhalf);
#pragma unroll
            for (int ti = 0; ti < 2; ++ti) {
                int j = 2 * wave + ti;
#pragma unroll
                for (int q = 0; q < 4; ++q) {
                    v16bf bb = load_b(whhb + (q * 256 + j * 16 + lmod) * HSZ
                                           + K0 + lhalf * 16);
                    g[ti][q] = wmma_bf16(a, bb, g[ti][q]);
                }
            }
        }

        // (2) + x@Wih^T + biases, nonlinearities, c/h update (registers)
        unsigned short hb[2][8];
#pragma unroll
        for (int ti = 0; ti < 2; ++ti) {
#pragma unroll
            for (int r = 0; r < 8; ++r) {
                int m = lhalf * 8 + r;
                float x0 = sh_x[m][0], x1 = sh_x[m][1], x2 = sh_x[m][2];
                float iv = g[ti][0][r] + bsum[ti][0] + x0 * wx[ti][0][0] + x1 * wx[ti][0][1] + x2 * wx[ti][0][2];
                float fv = g[ti][1][r] + bsum[ti][1] + x0 * wx[ti][1][0] + x1 * wx[ti][1][1] + x2 * wx[ti][1][2];
                float gv = g[ti][2][r] + bsum[ti][2] + x0 * wx[ti][2][0] + x1 * wx[ti][2][1] + x2 * wx[ti][2][2];
                float ov = g[ti][3][r] + bsum[ti][3] + x0 * wx[ti][3][0] + x1 * wx[ti][3][1] + x2 * wx[ti][3][2];
                float cv = sigf(fv) * cst[ti][r] + sigf(iv) * tanhfast(gv);
                cst[ti][r] = cv;
                hb[ti][r] = f2bf(sigf(ov) * tanhfast(cv));
            }
        }
        __syncthreads();  // all sh_h / sh_x reads of this step done
#pragma unroll
        for (int ti = 0; ti < 2; ++ti) {
            int nb = (2 * wave + ti) * 16;
#pragma unroll
            for (int r = 0; r < 8; ++r)
                sh_h[lhalf * 8 + r][nb + lmod] = hb[ti][r];
        }
        __syncthreads();  // new h visible

        // (3) z1 = relu(h @ Wo1^T + bo1) : M=16, N=128, K=256; wave w -> tile w
        v8f z = {0.f, 0.f, 0.f, 0.f, 0.f, 0.f, 0.f, 0.f};
        for (int k0 = 0; k0 < 8; ++k0) {
            int K0 = k0 * 32;
            v16bf a  = load_a(&sh_h[lmod][0], K0, lhalf);
            v16bf bb = load_b(wo1b + (wave * 16 + lmod) * HSZ + K0 + lhalf * 16);
            z = wmma_bf16(a, bb, z);
        }
#pragma unroll
        for (int r = 0; r < 8; ++r)
            sh_z1[lhalf * 8 + r][wave * 16 + lmod] = fmaxf(z[r] + bo1v, 0.0f);
        __syncthreads();

        // (4) y = z1 @ Wo2^T + bo2 (N=3): 48 lanes, write out and feed back as x
        if (tid < 48) {
            int m = tid / 3, o = tid - m * 3;
            float acc = sh_bo2v[o];
#pragma unroll 8
            for (int n = 0; n < 128; ++n) acc += sh_z1[m][n] * sh_wo2[o][n];
            out[((size_t)(row0 + m) * T + t) * 3 + o] = acc;
            sh_x[m][o] = acc;
        }
        __syncthreads();  // x(t+1) visible before next step's gate finishing
    }
}

extern "C" void kernel_launch(void* const* d_in, const int* in_sizes, int n_in,
                              void* d_out, int out_size, void* d_ws, size_t ws_size,
                              hipStream_t stream) {
    const float* meta = (const float*)d_in[0];
    const int*   tsl  = (const int*)  d_in[1];
    const float* W1   = (const float*)d_in[2];
    const float* b1   = (const float*)d_in[3];
    const float* W2   = (const float*)d_in[4];
    const float* b2   = (const float*)d_in[5];
    const float* Wih  = (const float*)d_in[6];
    const float* Whh  = (const float*)d_in[7];
    const float* bih  = (const float*)d_in[8];
    const float* bhh  = (const float*)d_in[9];
    const float* Wo1  = (const float*)d_in[10];
    const float* bo1  = (const float*)d_in[11];
    const float* Wo2  = (const float*)d_in[12];
    const float* bo2  = (const float*)d_in[13];

    unsigned short* ws = (unsigned short*)d_ws;  // bf16 weight cache: W2|Whh|Wo1
    const int tot = N2 + NH + NO;

    hipLaunchKernelGGL(conv_bf16_kernel, dim3((tot + 255) / 256), dim3(256), 0, stream,
                       W2, Whh, Wo1, ws);
    hipLaunchKernelGGL(dubins_lstm_kernel, dim3(4096 / 16), dim3(256), 0, stream,
                       meta, tsl, W1, b1, b2, Wih, bih, bhh, bo1, Wo2, bo2,
                       (const unsigned short*)ws,
                       (const unsigned short*)(ws + N2),
                       (const unsigned short*)(ws + N2 + NH),
                       (float*)d_out);
}
